// DKRZSeismologyLSTM_44195213475890
// MI455X (gfx1250) — matrix operand order repaired and checked by reading, not compiled, not run on hardware
//
#include <hip/hip_runtime.h>
#include <hip/hip_bf16.h>

typedef __attribute__((ext_vector_type(16))) _Float16 v16h;
typedef __attribute__((ext_vector_type(8)))  _Float16 v8h;
typedef __attribute__((ext_vector_type(8)))  float    v8f;

#define B_TOT  256
#define T_LEN  2048
#define IN_DIM 49
#define H_DIM  64
#define G_DIM  256   /* 4*H */
#define MB     16    /* batch rows per workgroup (WMMA M) */
#define HS     72    /* padded f16 row stride for x/h LDS tiles */

__device__ __forceinline__ float sig_f(float x) { return 1.0f / (1.0f + __expf(-x)); }
__device__ __forceinline__ float tanh_f(float x) { return 1.0f - 2.0f / (__expf(2.0f * x) + 1.0f); }

__launch_bounds__(256, 1)
__global__ void lstm2_persistent_wmma(const float* __restrict__ x,
                                      const float* __restrict__ W_ih1, const float* __restrict__ b_ih1,
                                      const float* __restrict__ W_hh1, const float* __restrict__ b_hh1,
                                      const float* __restrict__ W_ih2, const float* __restrict__ b_ih2,
                                      const float* __restrict__ W_hh2, const float* __restrict__ b_hh2,
                                      const float* __restrict__ W_out, const float* __restrict__ b_out,
                                      float* __restrict__ out)
{
    __shared__ __align__(16) _Float16 xs [MB * HS];
    __shared__ __align__(16) _Float16 h1s[MB * HS];
    __shared__ __align__(16) _Float16 h2s[MB * HS];
    __shared__ __align__(16) float gates[MB * G_DIM];
    __shared__ __align__(16) float c1s[MB * H_DIM];
    __shared__ __align__(16) float c2s[MB * H_DIM];
    __shared__ float bias1[G_DIM];
    __shared__ float bias2[G_DIM];
    __shared__ float woutS[H_DIM];
    __shared__ float red[256];

    const int tid  = threadIdx.x;
    const int lane = tid & 31;
    const int wv   = tid >> 5;          // wave 0..7, owns N-tiles 2*wv, 2*wv+1
    const int b0   = blockIdx.x * MB;   // batch slice base

    // ---------------- one-time init ----------------
    if (tid < G_DIM) {
        bias1[tid] = b_ih1[tid] + b_hh1[tid];
        bias2[tid] = b_ih2[tid] + b_hh2[tid];
    }
    if (tid < H_DIM) woutS[tid] = W_out[tid];
    for (int i = tid; i < MB * HS; i += 256) { h1s[i] = (_Float16)0.f; h2s[i] = (_Float16)0.f; }
    for (int i = tid; i < MB * H_DIM; i += 256) { c1s[i] = 0.f; c2s[i] = 0.f; }
    const float boutv = b_out[0];

    // ------- preload all weight B-fragments into registers (f16) -------
    // B fragment lane layout (16x16x32 f16, B is 32x16 = W^T slice):
    //   n = tile*16 + (lane&15), element j (0..15) holds K = ks*32 + (lane>>4)*16 + j
    v16h wf[4][2][2];   // [matrix][owned-tile][k-step]
    {
        const float* Wm[4] = { W_ih1, W_hh1, W_ih2, W_hh2 };
        const int    RS[4] = { IN_DIM, H_DIM, H_DIM, H_DIM };   // row stride
        const int    KL[4] = { IN_DIM, H_DIM, H_DIM, H_DIM };   // valid K (pad with 0)
#pragma unroll
        for (int m = 0; m < 4; ++m) {
#pragma unroll
            for (int ti = 0; ti < 2; ++ti) {
                const int n = (2 * wv + ti) * 16 + (lane & 15);
#pragma unroll
                for (int ks = 0; ks < 2; ++ks) {
                    const int kbase = ks * 32 + (lane >> 4) * 16;
                    v16h w;
#pragma unroll
                    for (int j = 0; j < 16; ++j) {
                        const int k = kbase + j;
                        w[j] = (k < KL[m]) ? (_Float16)Wm[m][n * RS[m] + k] : (_Float16)0.f;
                    }
                    wf[m][ti][ks] = w;
                }
            }
        }
    }
    __syncthreads();

    // A fragment lane constants (16x32 f16 A): row m = lane&15, K-half base kb = (lane>>4)*16
    const int am  = lane & 15;
    const int akb = (lane >> 4) * 16;
    const int dn  = lane & 15;          // D: column within tile
    const int dm0 = (lane >> 4) * 8;    // D: VGPR0 maps to row dm0
    const int eb  = tid >> 4;           // elementwise: batch row 0..15
    const int ej0 = (tid & 15) * 4;     // elementwise: hidden idx base

    for (int t = 0; t < T_LEN; ++t) {
        // ---------- stage x_t (f32 -> f16, pad 49->64), prefetch next ----------
        {
            const int bsb = tid >> 4;
            const int k0  = (tid & 15) * 4;
            const float* xrow = x + ((size_t)(b0 + bsb) * T_LEN + t) * IN_DIM;
#pragma unroll
            for (int q = 0; q < 4; ++q) {
                const int k = k0 + q;
                xs[bsb * HS + k] = (_Float16)((k < IN_DIM) ? xrow[k] : 0.f);
            }
            if (t + 1 < T_LEN && k0 < IN_DIM)
                __builtin_prefetch(xrow + IN_DIM + k0, 0, 0);   // global_prefetch_b8
        }
        __syncthreads();

        // ---------- GEMM1: gates = bias1 + x_t @ W_ih1^T + h1 @ W_hh1^T ----------
        {
            v16h ax[2], ah[2];
#pragma unroll
            for (int ks = 0; ks < 2; ++ks) {
                const int base = ks * 32;
                v8h xlo = *(const v8h*)&xs [am * HS + base + akb];
                v8h xhi = *(const v8h*)&xs [am * HS + base + 8 + akb];
                v8h hlo = *(const v8h*)&h1s[am * HS + base + akb];
                v8h hhi = *(const v8h*)&h1s[am * HS + base + 8 + akb];
#pragma unroll
                for (int e = 0; e < 8; ++e) {
                    ax[ks][e] = xlo[e]; ax[ks][8 + e] = xhi[e];
                    ah[ks][e] = hlo[e]; ah[ks][8 + e] = hhi[e];
                }
            }
#pragma unroll
            for (int ti = 0; ti < 2; ++ti) {
                const int n = (2 * wv + ti) * 16 + dn;
                const float bz = bias1[n];
                v8f acc;
#pragma unroll
                for (int v = 0; v < 8; ++v) acc[v] = bz;
                acc = __builtin_amdgcn_wmma_f32_16x16x32_f16(false, ax[0], false, wf[0][ti][0], (short)0, acc, false, false);
                acc = __builtin_amdgcn_wmma_f32_16x16x32_f16(false, ax[1], false, wf[0][ti][1], (short)0, acc, false, false);
                acc = __builtin_amdgcn_wmma_f32_16x16x32_f16(false, ah[0], false, wf[1][ti][0], (short)0, acc, false, false);
                acc = __builtin_amdgcn_wmma_f32_16x16x32_f16(false, ah[1], false, wf[1][ti][1], (short)0, acc, false, false);
#pragma unroll
                for (int v = 0; v < 8; ++v) gates[(dm0 + v) * G_DIM + n] = acc[v];
            }
        }
        __syncthreads();

        // ---------- elementwise cell 1 ----------
        {
#pragma unroll
            for (int q = 0; q < 4; ++q) {
                const int j = ej0 + q;
                const float iv = sig_f (gates[eb * G_DIM +        j]);
                const float fv = sig_f (gates[eb * G_DIM +  64 + j]);
                const float gv = tanh_f(gates[eb * G_DIM + 128 + j]);
                const float ov = sig_f (gates[eb * G_DIM + 192 + j]);
                const float c  = fv * c1s[eb * H_DIM + j] + iv * gv;
                c1s[eb * H_DIM + j] = c;
                h1s[eb * HS + j] = (_Float16)(ov * tanh_f(c));
            }
        }
        __syncthreads();

        // ---------- GEMM2: gates = bias2 + h1 @ W_ih2^T + h2 @ W_hh2^T ----------
        {
            v16h a1[2], a2[2];
#pragma unroll
            for (int ks = 0; ks < 2; ++ks) {
                const int base = ks * 32;
                v8h lo1 = *(const v8h*)&h1s[am * HS + base + akb];
                v8h hi1 = *(const v8h*)&h1s[am * HS + base + 8 + akb];
                v8h lo2 = *(const v8h*)&h2s[am * HS + base + akb];
                v8h hi2 = *(const v8h*)&h2s[am * HS + base + 8 + akb];
#pragma unroll
                for (int e = 0; e < 8; ++e) {
                    a1[ks][e] = lo1[e]; a1[ks][8 + e] = hi1[e];
                    a2[ks][e] = lo2[e]; a2[ks][8 + e] = hi2[e];
                }
            }
#pragma unroll
            for (int ti = 0; ti < 2; ++ti) {
                const int n = (2 * wv + ti) * 16 + dn;
                const float bz = bias2[n];
                v8f acc;
#pragma unroll
                for (int v = 0; v < 8; ++v) acc[v] = bz;
                acc = __builtin_amdgcn_wmma_f32_16x16x32_f16(false, a1[0], false, wf[2][ti][0], (short)0, acc, false, false);
                acc = __builtin_amdgcn_wmma_f32_16x16x32_f16(false, a1[1], false, wf[2][ti][1], (short)0, acc, false, false);
                acc = __builtin_amdgcn_wmma_f32_16x16x32_f16(false, a2[0], false, wf[3][ti][0], (short)0, acc, false, false);
                acc = __builtin_amdgcn_wmma_f32_16x16x32_f16(false, a2[1], false, wf[3][ti][1], (short)0, acc, false, false);
#pragma unroll
                for (int v = 0; v < 8; ++v) gates[(dm0 + v) * G_DIM + n] = acc[v];
            }
        }
        __syncthreads();

        // ---------- elementwise cell 2 + output partial ----------
        {
            float partial = 0.f;
#pragma unroll
            for (int q = 0; q < 4; ++q) {
                const int j = ej0 + q;
                const float iv = sig_f (gates[eb * G_DIM +        j]);
                const float fv = sig_f (gates[eb * G_DIM +  64 + j]);
                const float gv = tanh_f(gates[eb * G_DIM + 128 + j]);
                const float ov = sig_f (gates[eb * G_DIM + 192 + j]);
                const float c  = fv * c2s[eb * H_DIM + j] + iv * gv;
                c2s[eb * H_DIM + j] = c;
                const float hn = ov * tanh_f(c);
                h2s[eb * HS + j] = (_Float16)hn;
                partial += hn * woutS[j];
            }
            red[tid] = partial;   // red[b*16 + jgroup]
        }
        __syncthreads();

        if (tid < MB) {
            float s = boutv;
#pragma unroll
            for (int q = 0; q < 16; ++q) s += red[tid * 16 + q];
            out[(size_t)(b0 + tid) * T_LEN + t] = s;
        }
        __syncthreads();
    }
}

extern "C" void kernel_launch(void* const* d_in, const int* in_sizes, int n_in,
                              void* d_out, int out_size, void* d_ws, size_t ws_size,
                              hipStream_t stream) {
    (void)in_sizes; (void)n_in; (void)out_size; (void)d_ws; (void)ws_size;
    const float* x     = (const float*)d_in[0];
    const float* W_ih1 = (const float*)d_in[1];
    const float* b_ih1 = (const float*)d_in[2];
    const float* W_hh1 = (const float*)d_in[3];
    const float* b_hh1 = (const float*)d_in[4];
    const float* W_ih2 = (const float*)d_in[5];
    const float* b_ih2 = (const float*)d_in[6];
    const float* W_hh2 = (const float*)d_in[7];
    const float* b_hh2 = (const float*)d_in[8];
    const float* W_out = (const float*)d_in[9];
    const float* b_out = (const float*)d_in[10];
    float* out = (float*)d_out;

    dim3 grid(B_TOT / MB);   // 16 workgroups, one 16-row batch slice each
    dim3 block(256);         // 8 waves (wave32)
    lstm2_persistent_wmma<<<grid, block, 0, stream>>>(x, W_ih1, b_ih1, W_hh1, b_hh1,
                                                      W_ih2, b_ih2, W_hh2, b_hh2,
                                                      W_out, b_out, out);
}
